// LocationSensitiveAttention_34437047779865
// MI455X (gfx1250) — matrix operand order repaired and checked by reading, not compiled, not run on hardware
//
#include <hip/hip_runtime.h>
#include <hip/hip_bf16.h>

#define BB   32
#define TT   4096
#define AA   128
#define EE   512
#define DD   1024

typedef __attribute__((ext_vector_type(16))) __bf16 v16bf;
typedef __attribute__((ext_vector_type(8)))  float  v8f;

__device__ __forceinline__ __bf16 f2bf(float f) {
    unsigned u = __builtin_bit_cast(unsigned, f);
    // round-to-nearest-even fp32 -> bf16
    unsigned r = u + 0x7FFFu + ((u >> 16) & 1u);
    unsigned short h = (unsigned short)(r >> 16);
    return __builtin_bit_cast(__bf16, h);
}

// ---------------------------------------------------------------------------
// pq[b,a] = sum_d query[b,d] * Wq[d,a]      (32 blocks x 128 threads)
// ---------------------------------------------------------------------------
__global__ void pq_kernel(const float* __restrict__ q,
                          const float* __restrict__ Wq,
                          float* __restrict__ pq) {
    int b = blockIdx.x;
    int a = threadIdx.x;
    const float* qr = q + (size_t)b * DD;
    float s = 0.f;
    #pragma unroll 4
    for (int d = 0; d < DD; ++d) s = fmaf(qr[d], Wq[(size_t)d * AA + a], s);
    pq[b * AA + a] = s;
}

// ---------------------------------------------------------------------------
// Pack Wm [512,128] fp32 row-major into bf16 WMMA B fragments.
// Layout: fragment (ntile 0..7, kstep 0..15) -> 32 lanes x 16 halves, where
// lane L holds column N = ntile*16 + (L%16), elements j=0..15 are
// K = kstep*32 + (j/8)*16 + (L/16)*8 + (j%8)  (CDNA5 16-bit B-matrix layout).
// Total 8*16*32*16 = 65536 bf16 = 128 KB.
// ---------------------------------------------------------------------------
__global__ void packB_kernel(const float* __restrict__ Wm,
                             __bf16* __restrict__ packB) {
    int idx   = blockIdx.x * 256 + threadIdx.x;      // 0 .. 65535
    int j     = idx & 15;
    int lane  = (idx >> 4) & 31;
    int frag  = idx >> 9;                            // ntile*16 + kstep
    int nt    = frag >> 4;
    int ks    = frag & 15;
    int K     = ks * 32 + ((j >> 3) << 4) + (lane >> 4) * 8 + (j & 7);
    int N     = nt * 16 + (lane & 15);
    packB[idx] = f2bf(Wm[(size_t)K * AA + N]);
}

// ---------------------------------------------------------------------------
// Fused pm-GEMM + tanh + dot(v) -> scores.
// One wave handles a 16-row T tile for one batch b:
//   for kstep(16): build A frag from memory (fp32 -> bf16), then 8 WMMAs
//   (one per 16-col N tile) into 8 f32 accumulators.
// Epilogue: score[m] = sum_n tanh(pm + pq) * v, shuffle-reduced over N lanes.
// 128 threads/block = 4 waves; grid = 32*256/4 = 2048 blocks.
// ---------------------------------------------------------------------------
__global__ void scores_kernel(const float* __restrict__ mem,    // [B,T,E]
                              const float* __restrict__ pq,     // [B,A]
                              const __bf16* __restrict__ packB, // fragments
                              const float* __restrict__ vv,     // [A]
                              float* __restrict__ scoresOut) {  // [B,T]
    const int lane = threadIdx.x & 31;
    const int wave = threadIdx.x >> 5;
    const int tile = blockIdx.x * 4 + wave;          // 0 .. 8191
    const int b    = tile >> 8;                      // T/16 = 256 tiles per b
    const int t0   = (tile & 255) << 4;

    const int half = lane >> 4;                      // 0 or 1
    const int col  = lane & 15;

    const float* arow = mem + ((size_t)b * TT + t0 + col) * EE;

    v8f c[8];
    #pragma unroll
    for (int n = 0; n < 8; ++n) c[n] = (v8f){0.f,0.f,0.f,0.f,0.f,0.f,0.f,0.f};

    for (int ks = 0; ks < 16; ++ks) {
        const int kbase = ks * 32;
        // A fragment: lanes 0-15 hold K = kbase+{0..7, 16..23},
        //             lanes 16-31 hold K = kbase+{8..15, 24..31}
        const float4* p0 = reinterpret_cast<const float4*>(arow + kbase + half * 8);
        const float4* p1 = reinterpret_cast<const float4*>(arow + kbase + 16 + half * 8);
        float4 q0 = p0[0], q1 = p0[1];
        float4 q2 = p1[0], q3 = p1[1];
        v16bf a;
        a[0]=f2bf(q0.x); a[1]=f2bf(q0.y); a[2]=f2bf(q0.z); a[3]=f2bf(q0.w);
        a[4]=f2bf(q1.x); a[5]=f2bf(q1.y); a[6]=f2bf(q1.z); a[7]=f2bf(q1.w);
        a[8]=f2bf(q2.x); a[9]=f2bf(q2.y); a[10]=f2bf(q2.z); a[11]=f2bf(q2.w);
        a[12]=f2bf(q3.x); a[13]=f2bf(q3.y); a[14]=f2bf(q3.z); a[15]=f2bf(q3.w);

        #pragma unroll
        for (int n = 0; n < 8; ++n) {
            const v16bf bfrag =
                *reinterpret_cast<const v16bf*>(packB + ((size_t)((n * 16 + ks) * 32 + lane) << 4));
            c[n] = __builtin_amdgcn_wmma_f32_16x16x32_bf16(
                       false, a, false, bfrag, (short)0, c[n], false, false);
        }
    }

    // Epilogue: lane holds D rows M = r + 8*half, col N = n*16 + col.
    float sacc[8];
    #pragma unroll
    for (int r = 0; r < 8; ++r) sacc[r] = 0.f;

    #pragma unroll
    for (int n = 0; n < 8; ++n) {
        const float pqn = pq[b * AA + n * 16 + col];
        const float vn  = vv[n * 16 + col];
        #pragma unroll
        for (int r = 0; r < 8; ++r)
            sacc[r] += tanhf(c[n][r] + pqn) * vn;
    }

    // Reduce over the 16 N-lanes within each half (xor offsets 1,2,4,8 stay in-half)
    #pragma unroll
    for (int r = 0; r < 8; ++r) {
        #pragma unroll
        for (int o = 1; o < 16; o <<= 1)
            sacc[r] += __shfl_xor(sacc[r], o, 32);
    }

    if (col == 0) {
        float* out = scoresOut + (size_t)b * TT + t0 + half * 8;
        #pragma unroll
        for (int r = 0; r < 8; ++r) out[r] = sacc[r];
    }
}

// ---------------------------------------------------------------------------
// In-place softmax over T per batch row (32 blocks x 256 threads).
// ---------------------------------------------------------------------------
__global__ void softmax_kernel(float* __restrict__ sc) {
    const int b = blockIdx.x;
    float* row = sc + (size_t)b * TT;
    __shared__ float red[8];
    const int tid = threadIdx.x, lane = tid & 31, w = tid >> 5;

    float m = -1e30f;
    for (int t = tid; t < TT; t += 256) m = fmaxf(m, row[t]);
    #pragma unroll
    for (int o = 16; o > 0; o >>= 1) m = fmaxf(m, __shfl_xor(m, o, 32));
    if (lane == 0) red[w] = m;
    __syncthreads();
    float bm = red[0];
    #pragma unroll
    for (int i = 1; i < 8; ++i) bm = fmaxf(bm, red[i]);

    float s = 0.f;
    for (int t = tid; t < TT; t += 256) {
        float e = __expf(row[t] - bm);
        row[t] = e;
        s += e;
    }
    #pragma unroll
    for (int o = 16; o > 0; o >>= 1) s += __shfl_xor(s, o, 32);
    __syncthreads();
    if (lane == 0) red[w] = s;
    __syncthreads();
    float bs = 0.f;
    #pragma unroll
    for (int i = 0; i < 8; ++i) bs += red[i];
    const float inv = 1.0f / bs;
    for (int t = tid; t < TT; t += 256) row[t] *= inv;
}

// ---------------------------------------------------------------------------
// context[b,e] = sum_t align[b,t] * memory[b,t,e]
// grid (4, 32) x 128 threads; coalesced over e, streams memory once.
// ---------------------------------------------------------------------------
__global__ void context_kernel(const float* __restrict__ mem,
                               const float* __restrict__ al,
                               float* __restrict__ ctx) {
    const int b = blockIdx.y;
    const int e = blockIdx.x * 128 + threadIdx.x;
    const float* mcol = mem + (size_t)b * TT * EE + e;
    const float* arow = al + (size_t)b * TT;
    float s = 0.f;
    for (int t = 0; t < TT; t += 4) {
        float4 wv = *reinterpret_cast<const float4*>(arow + t);
        s = fmaf(wv.x, mcol[(size_t)(t + 0) * EE], s);
        s = fmaf(wv.y, mcol[(size_t)(t + 1) * EE], s);
        s = fmaf(wv.z, mcol[(size_t)(t + 2) * EE], s);
        s = fmaf(wv.w, mcol[(size_t)(t + 3) * EE], s);
    }
    ctx[b * EE + e] = s;
}

// ---------------------------------------------------------------------------
extern "C" void kernel_launch(void* const* d_in, const int* in_sizes, int n_in,
                              void* d_out, int out_size, void* d_ws, size_t ws_size,
                              hipStream_t stream) {
    const float* query  = (const float*)d_in[0];   // [32,1024]
    const float* memory = (const float*)d_in[1];   // [32,4096,512]
    // d_in[2] = attention_weights_cum : unused by the reference math
    const float* Wq     = (const float*)d_in[3];   // [1024,128]
    const float* Wm     = (const float*)d_in[4];   // [512,128]
    const float* v      = (const float*)d_in[5];   // [128]

    float* ctx    = (float*)d_out;                 // [32,512]
    float* aligns = (float*)d_out + BB * EE;       // [32,4096]

    float*  pq    = (float*)d_ws;                           // 16 KB
    __bf16* packB = (__bf16*)((char*)d_ws + BB * AA * 4);   // 128 KB

    pq_kernel<<<BB, AA, 0, stream>>>(query, Wq, pq);
    packB_kernel<<<256, 256, 0, stream>>>(Wm, packB);
    scores_kernel<<<(BB * (TT / 16)) / 4, 128, 0, stream>>>(memory, pq, packB, v, aligns);
    softmax_kernel<<<BB, 256, 0, stream>>>(aligns);
    context_kernel<<<dim3(EE / 128, BB), 128, 0, stream>>>(memory, aligns, ctx);
}
